// StandardAttention_13932873908486
// MI455X (gfx1250) — matrix operand order repaired and checked by reading, not compiled, not run on hardware
//
#include <hip/hip_runtime.h>
#include <hip/hip_bf16.h>
#include <math.h>

typedef __bf16 bf16;
typedef __bf16 v16bf __attribute__((ext_vector_type(16)));
typedef float  v8f   __attribute__((ext_vector_type(8)));
typedef unsigned int u32x4 __attribute__((ext_vector_type(4)));
typedef unsigned int v4u  __attribute__((ext_vector_type(4)));
typedef int          v8i  __attribute__((ext_vector_type(8)));
typedef int          v4i  __attribute__((ext_vector_type(4)));

#define NB      2
#define SEQ     2048
#define HID     2048
#define NHEADS  16
#define NKVH    4
#define HDIM    128
#define MTOK    (NB*SEQ)
#define KVDIM   (NKVH*HDIM)

#if __has_builtin(__builtin_amdgcn_tensor_load_to_lds)
#define HAVE_TDM 1
#else
#define HAVE_TDM 0
#endif

__device__ __forceinline__ v8f vzero8() {
    v8f z;
    for (int i = 0; i < 8; ++i) z[i] = 0.0f;
    return z;
}

// Build a 16-element bf16 fragment from two 16-byte chunks.
__device__ __forceinline__ v16bf make_frag(const bf16* p0, const bf16* p1) {
    v16bf f;
    u32x4* d = reinterpret_cast<u32x4*>(&f);
    d[0] = *reinterpret_cast<const u32x4*>(p0);
    d[1] = *reinterpret_cast<const u32x4*>(p1);
    return f;
}

__device__ __forceinline__ v8f wmma_bf16(v16bf a, v16bf b, v8f c) {
    return __builtin_amdgcn_wmma_f32_16x16x32_bf16(
        /*neg_a=*/false, a, /*neg_b=*/false, b,
        /*c_mod=*/(short)0, c, /*reuse_a=*/false, /*reuse_b=*/false);
}

__device__ __forceinline__ float rmax16(float x) {
    x = fmaxf(x, __shfl_xor(x, 1, 32));
    x = fmaxf(x, __shfl_xor(x, 2, 32));
    x = fmaxf(x, __shfl_xor(x, 4, 32));
    x = fmaxf(x, __shfl_xor(x, 8, 32));
    return x;
}
__device__ __forceinline__ float rsum16(float x) {
    x += __shfl_xor(x, 1, 32);
    x += __shfl_xor(x, 2, 32);
    x += __shfl_xor(x, 4, 32);
    x += __shfl_xor(x, 8, 32);
    return x;
}

#if HAVE_TDM
// TDM: DMA one [128 rows x 32 halves] bf16 tile from a row-major [rows][K]
// matrix into LDS with a 40-half (80B) padded row pitch:
//   pad_interval=3 -> pad after every 16 DWORDs (64B = 32 halves)
//   pad_amount=3   -> insert 4 DWORDs (16B = 8 halves) of padding
__device__ __forceinline__ void tdm_load_tile(unsigned lds_byte, const bf16* gsrc,
                                              int K, int nrows) {
    unsigned long long ga = (unsigned long long)(size_t)gsrc;
    v4u g0;
    g0[0] = 1u;                                   // count=1, user mode
    g0[1] = lds_byte;                             // lds_addr
    g0[2] = (unsigned)(ga & 0xffffffffull);       // global_addr[31:0]
    g0[3] = (unsigned)((ga >> 32) & 0x1ffffffull) // global_addr[56:32]
            | (2u << 30);                         // type=2 (image)
    v8i g1;
    g1[0] = (1 << 16)            // data_size = 1 (2 bytes)
          | (1 << 20)            // pad_enable
          | (3 << 22)            // pad_interval: 16 DWORDs
          | (3 << 25);           // pad_amount:   4 DWORDs
    g1[1] = (int)(((unsigned)K & 0xffffu) << 16);                 // tensor_dim0[15:0]
    g1[2] = (int)((((unsigned)K >> 16) & 0xffffu)
          | (((unsigned)nrows & 0xffffu) << 16));                 // dim0 hi | dim1 lo
    g1[3] = (int)((((unsigned)nrows >> 16) & 0xffffu)
          | (32u << 16));                                         // dim1 hi | tile_dim0=32
    g1[4] = 128;                                                  // tile_dim1=128, tile_dim2=0
    g1[5] = K;                                                    // tensor_dim0_stride lo
    g1[6] = 0;                                                    // stride0 hi | stride1 lo
    g1[7] = 0;                                                    // stride1 hi
    v4i g2; g2[0] = 1; g2[1] = 0; g2[2] = 0; g2[3] = 0;           // tensor_dim2=1
    v4i g3; g3[0] = 0; g3[1] = 0; g3[2] = 0; g3[3] = 0;
#if __clang_major__ >= 23
    v8i g4; for (int i = 0; i < 8; ++i) g4[i] = 0;
    __builtin_amdgcn_tensor_load_to_lds(g0, g1, g2, g3, g4, 0);
#else
    __builtin_amdgcn_tensor_load_to_lds(g0, g1, g2, g3, 0);
#endif
}
#endif

// ---------------- fp32 -> bf16 conversion ----------------
__global__ __launch_bounds__(256) void f32_to_bf16_kernel(
        const float* __restrict__ in, bf16* __restrict__ out, int n) {
    int i = (blockIdx.x * blockDim.x + threadIdx.x) * 4;
    if (i + 3 >= n) return;
    float4 v = *reinterpret_cast<const float4*>(in + i);
    union { bf16 h[4]; uint2 u; } pk;
    pk.h[0] = (bf16)v.x; pk.h[1] = (bf16)v.y;
    pk.h[2] = (bf16)v.z; pk.h[3] = (bf16)v.w;
    *reinterpret_cast<uint2*>(out + i) = pk.u;
}

// ---------------- W[K][N] fp32 -> WT[N][K] bf16 ----------------
__global__ __launch_bounds__(256) void transpose_w_kernel(
        const float* __restrict__ W, bf16* __restrict__ WT, int K, int N) {
    __shared__ float tile[32][33];
    const int n0 = blockIdx.x * 32, k0 = blockIdx.y * 32;
    const int tx = threadIdx.x, ty = threadIdx.y;
#pragma unroll
    for (int i = 0; i < 4; ++i)
        tile[ty + i * 8][tx] = W[(size_t)(k0 + ty + i * 8) * N + n0 + tx];
    __syncthreads();
#pragma unroll
    for (int i = 0; i < 4; ++i)
        WT[(size_t)(n0 + ty + i * 8) * K + k0 + tx] = (bf16)tile[tx][ty + i * 8];
}

// ---------------- V [bh][S][d] -> Vt [bh][d][S] (bf16) ----------------
__global__ __launch_bounds__(256) void transpose_v_kernel(
        const bf16* __restrict__ V, bf16* __restrict__ Vt) {
    __shared__ bf16 tile[32][33];
    const int bh = blockIdx.z;
    const int d0 = blockIdx.x * 32, s0 = blockIdx.y * 32;
    const bf16* src = V + (size_t)bh * SEQ * HDIM;
    bf16* dst = Vt + (size_t)bh * SEQ * HDIM;
    const int tx = threadIdx.x, ty = threadIdx.y;
#pragma unroll
    for (int i = 0; i < 4; ++i)
        tile[ty + i * 8][tx] = src[(size_t)(s0 + ty + i * 8) * HDIM + d0 + tx];
    __syncthreads();
#pragma unroll
    for (int i = 0; i < 4; ++i)
        dst[(size_t)(d0 + ty + i * 8) * SEQ + s0 + tx] = tile[tx][ty + i * 8];
}

// ---------------- RoPE in place on [B][nh][S][HDIM] bf16 ----------------
__global__ __launch_bounds__(256) void rope_kernel(bf16* __restrict__ q, int total) {
    int idx = blockIdx.x * blockDim.x + threadIdx.x;
    if (idx >= total) return;
    const int i  = idx & 63;
    const int s  = (idx >> 6) & (SEQ - 1);
    const int bh = idx >> 17;
    float f = (float)s * exp2f((float)i * (-19.9315685693241741f / 64.0f));
    float cs = cosf(f), sn = sinf(f);
    size_t base = ((size_t)bh * SEQ + s) * HDIM;
    float x1 = (float)q[base + i], x2 = (float)q[base + i + 64];
    q[base + i]      = (bf16)(x1 * cs - x2 * sn);
    q[base + i + 64] = (bf16)(x2 * cs + x1 * sn);
}

// ---------------- tiled WMMA GEMM: C = X(MxK) * WT(NxK)^T + bias ----------------
// A/B tiles staged by the Tensor Data Mover (one descriptor per 128x32 tile),
// double-buffered; 8 waves x (64x32) WMMA tiles.
// OUT_MODE 0: write bf16 scattered to [b][h][S][HDIM];  OUT_MODE 1: f32 [M][N]
template <int OUT_MODE>
__global__ __launch_bounds__(256) void gemm_bf16_wmma(
        const bf16* __restrict__ X, const bf16* __restrict__ WT,
        const float* __restrict__ bias, void* __restrict__ outp,
        int M, int N, int K, int numH) {
    __shared__ __attribute__((aligned(16))) bf16 As[2][128 * 40];
    __shared__ __attribute__((aligned(16))) bf16 Bs[2][128 * 40];

    const int t = threadIdx.x;
    const int wave = t >> 5, lane = t & 31;
    const int lm = lane & 15, hi = lane >> 4;
    const int kb = hi * 8, kb16 = hi * 16;
    const int wm = (wave & 1) * 64, wn = (wave >> 1) * 32;
    const int mbase = blockIdx.y * 128, nbase = blockIdx.x * 128;

    v8f acc[4][2];
#pragma unroll
    for (int mt = 0; mt < 4; ++mt) {
        acc[mt][0] = vzero8();
        acc[mt][1] = vzero8();
    }
    const int KT = K >> 5;

#if HAVE_TDM
    const bf16* gAt = X + (size_t)mbase * K;
    const bf16* gBt = WT + (size_t)nbase * K;
    const unsigned ldsA0 = (unsigned)(size_t)&As[0][0];
    const unsigned ldsA1 = (unsigned)(size_t)&As[1][0];
    const unsigned ldsB0 = (unsigned)(size_t)&Bs[0][0];
    const unsigned ldsB1 = (unsigned)(size_t)&Bs[1][0];

    if (wave == 0) {
        tdm_load_tile(ldsA0, gAt, K, M);
        tdm_load_tile(ldsB0, gBt, K, N);
    }
    for (int kt = 0; kt < KT; ++kt) {
        const int cur = kt & 1;
        if (wave == 0) __builtin_amdgcn_s_wait_tensorcnt(0);
        __syncthreads();   // buffer `cur` DMA complete, prior reads of `cur^1` done
        if (wave == 0 && kt + 1 < KT) {
            tdm_load_tile(cur ? ldsA0 : ldsA1, gAt + (size_t)(kt + 1) * 32, K, M);
            tdm_load_tile(cur ? ldsB0 : ldsB1, gBt + (size_t)(kt + 1) * 32, K, N);
        }
        v16bf bf0 = make_frag(&Bs[cur][(wn + lm) * 40 + kb],
                              &Bs[cur][(wn + lm) * 40 + kb + 16]);
        v16bf bf1 = make_frag(&Bs[cur][(wn + 16 + lm) * 40 + kb],
                              &Bs[cur][(wn + 16 + lm) * 40 + kb + 16]);
#pragma unroll
        for (int mt = 0; mt < 4; ++mt) {
            v16bf af = make_frag(&As[cur][(wm + mt * 16 + lm) * 40 + kb],
                                 &As[cur][(wm + mt * 16 + lm) * 40 + kb + 16]);
            acc[mt][0] = wmma_bf16(af, bf0, acc[mt][0]);
            acc[mt][1] = wmma_bf16(af, bf1, acc[mt][1]);
        }
    }
#else
    const int lrow = t >> 1;
    const int lcol = (t & 1) * 16;
    const bf16* gA = X + (size_t)(mbase + lrow) * K + lcol;
    const bf16* gB = WT + (size_t)(nbase + lrow) * K + lcol;
    auto stage = [&](int buf, int k0) {
        const u32x4* a = reinterpret_cast<const u32x4*>(gA + k0);
        const u32x4* b = reinterpret_cast<const u32x4*>(gB + k0);
        u32x4* da = reinterpret_cast<u32x4*>(&As[buf][lrow * 40 + lcol]);
        u32x4* db = reinterpret_cast<u32x4*>(&Bs[buf][lrow * 40 + lcol]);
        da[0] = a[0]; da[1] = a[1];
        db[0] = b[0]; db[1] = b[1];
    };
    stage(0, 0);
    __syncthreads();
    for (int kt = 0; kt < KT; ++kt) {
        const int cur = kt & 1;
        if (kt + 1 < KT) stage(cur ^ 1, (kt + 1) * 32);
        v16bf bf0 = make_frag(&Bs[cur][(wn + lm) * 40 + kb],
                              &Bs[cur][(wn + lm) * 40 + kb + 16]);
        v16bf bf1 = make_frag(&Bs[cur][(wn + 16 + lm) * 40 + kb],
                              &Bs[cur][(wn + 16 + lm) * 40 + kb + 16]);
#pragma unroll
        for (int mt = 0; mt < 4; ++mt) {
            v16bf af = make_frag(&As[cur][(wm + mt * 16 + lm) * 40 + kb],
                                 &As[cur][(wm + mt * 16 + lm) * 40 + kb + 16]);
            acc[mt][0] = wmma_bf16(af, bf0, acc[mt][0]);
            acc[mt][1] = wmma_bf16(af, bf1, acc[mt][1]);
        }
        __syncthreads();
    }
#endif

#pragma unroll
    for (int mt = 0; mt < 4; ++mt) {
#pragma unroll
        for (int nt = 0; nt < 2; ++nt) {
            const int ng = nbase + wn + nt * 16 + lm;
            const float bia = bias[ng];
#pragma unroll
            for (int v = 0; v < 8; ++v) {
                const int mg = mbase + wm + mt * 16 + v + 8 * hi;
                const float c = acc[mt][nt][v] + bia;
                if (OUT_MODE == 0) {
                    const int bb = mg >> 11, ss = mg & (SEQ - 1);
                    const int hh = ng >> 7, dd = ng & (HDIM - 1);
                    reinterpret_cast<bf16*>(outp)
                        [(((size_t)bb * numH + hh) * SEQ + ss) * HDIM + dd] = (bf16)c;
                } else {
                    reinterpret_cast<float*>(outp)[(size_t)mg * N + ng] = c;
                }
            }
        }
    }
}

// ---------------- flash attention with WMMA ----------------
// grid (SEQ/64, NHEADS, NB), 128 threads (4 waves x 16 query rows)
__global__ __launch_bounds__(128) void flash_attn_kernel(
        const bf16* __restrict__ Q, const bf16* __restrict__ Kc,
        const bf16* __restrict__ Vt, bf16* __restrict__ AO) {
    __shared__ __attribute__((aligned(16))) bf16 Plds[4][16][72];
    const int wave = threadIdx.x >> 5, lane = threadIdx.x & 31;
    const int lm = lane & 15, hi = lane >> 4;
    const int kb = hi * 8, kb16 = hi * 16;
    const int b = blockIdx.z, h = blockIdx.y;
    const int kh = h >> 2;  // NHEADS/NKVH = 4
    const int qbase = blockIdx.x * 64 + wave * 16;

    const bf16* Qb = Q  + (((size_t)b * NHEADS + h) * SEQ + qbase) * HDIM;
    const bf16* Kb = Kc + ((size_t)b * NKVH + kh) * SEQ * HDIM;   // [S][d]
    const bf16* Vb = Vt + ((size_t)b * NKVH + kh) * SEQ * HDIM;   // [d][S]

    v16bf qf[4];
#pragma unroll
    for (int tq = 0; tq < 4; ++tq)
        qf[tq] = make_frag(Qb + lm * HDIM + tq * 32 + kb,
                           Qb + lm * HDIM + tq * 32 + kb + 16);

    float mi[8], li[8];
    v8f o[8];
#pragma unroll
    for (int v = 0; v < 8; ++v) { mi[v] = -1e30f; li[v] = 0.0f; }
#pragma unroll
    for (int dj = 0; dj < 8; ++dj) o[dj] = vzero8();

    const float scl = 0.08838834764831845f;  // 1/sqrt(128)
    const int nb = blockIdx.x + 1;           // causal upper bound

    for (int blk = 0; blk < nb; ++blk) {
        const int kv0 = blk * 64;
        // prefetch next KV block's K rows (global_prefetch_b8)
        if (blk + 1 < nb) {
            const bf16* kn = Kb + (size_t)(kv0 + 64 + lane) * HDIM;
            __builtin_prefetch(kn, 0, 1);
            __builtin_prefetch(kn + 64, 0, 1);
            __builtin_prefetch(kn + 32 * HDIM, 0, 1);
            __builtin_prefetch(kn + 32 * HDIM + 64, 0, 1);
        }
        // ---- scores S = Q K^T ----
        v8f sc[4];
#pragma unroll
        for (int j = 0; j < 4; ++j) sc[j] = vzero8();
#pragma unroll
        for (int j = 0; j < 4; ++j) {
            const bf16* kr = Kb + (size_t)(kv0 + j * 16 + lm) * HDIM;
#pragma unroll
            for (int tq = 0; tq < 4; ++tq) {
                v16bf kf = make_frag(kr + tq * 32 + kb, kr + tq * 32 + kb + 16);
                sc[j] = wmma_bf16(qf[tq], kf, sc[j]);
            }
        }
        // ---- scale + causal mask ----
        const bool need_mask = (kv0 + 63) > qbase;
        float sfl[4][8];
#pragma unroll
        for (int j = 0; j < 4; ++j) {
            const int kg = kv0 + j * 16 + lm;
#pragma unroll
            for (int v = 0; v < 8; ++v) {
                float s = sc[j][v] * scl;
                if (need_mask && kg > qbase + v + 8 * hi) s = -1e30f;
                sfl[j][v] = s;
            }
        }
        // ---- online softmax ----
        float alpha[8];
#pragma unroll
        for (int v = 0; v < 8; ++v) {
            float r = fmaxf(fmaxf(sfl[0][v], sfl[1][v]), fmaxf(sfl[2][v], sfl[3][v]));
            r = rmax16(r);
            const float m2 = fmaxf(mi[v], r);
            alpha[v] = __expf(mi[v] - m2);
            mi[v] = m2;
        }
        float ps[8];
#pragma unroll
        for (int v = 0; v < 8; ++v) ps[v] = 0.0f;
#pragma unroll
        for (int j = 0; j < 4; ++j)
#pragma unroll
            for (int v = 0; v < 8; ++v) {
                const float p = __expf(sfl[j][v] - mi[v]);
                ps[v] += p;
                Plds[wave][v + 8 * hi][j * 16 + lm] = (bf16)p;
            }
#pragma unroll
        for (int v = 0; v < 8; ++v) {
            li[v] = li[v] * alpha[v] + rsum16(ps[v]);
        }
#pragma unroll
        for (int dj = 0; dj < 8; ++dj)
#pragma unroll
            for (int v = 0; v < 8; ++v) o[dj][v] *= alpha[v];

        // Per-wave LDS ordering: wait for P stores before A-layout reads.
        __asm__ volatile("s_wait_dscnt 0x0" ::: "memory");

        // ---- O += P V ----
#pragma unroll
        for (int kk = 0; kk < 2; ++kk) {
            v16bf pf = make_frag(&Plds[wave][lm][kk * 32 + kb],
                                 &Plds[wave][lm][kk * 32 + kb + 16]);
#pragma unroll
            for (int dj = 0; dj < 8; ++dj) {
                const bf16* vr = Vb + (size_t)(dj * 16 + lm) * SEQ + kv0 + kk * 32;
                v16bf vf = make_frag(vr + kb, vr + kb + 16);
                o[dj] = wmma_bf16(pf, vf, o[dj]);
            }
        }
    }

#pragma unroll
    for (int dj = 0; dj < 8; ++dj) {
#pragma unroll
        for (int v = 0; v < 8; ++v) {
            const float r = o[dj][v] / li[v];
            const int qg = qbase + v + 8 * hi;
            AO[((size_t)b * SEQ + qg) * HID + h * HDIM + dj * 16 + lm] = (bf16)r;
        }
    }
}

extern "C" void kernel_launch(void* const* d_in, const int* in_sizes, int n_in,
                              void* d_out, int out_size, void* d_ws, size_t ws_size,
                              hipStream_t stream) {
    (void)in_sizes; (void)n_in; (void)out_size; (void)ws_size;
    const float* hs = (const float*)d_in[0];
    // d_in[1] = attention_mask: analytic causal mask, applied analytically.
    const float* Wq = (const float*)d_in[2];
    const float* bq = (const float*)d_in[3];
    const float* Wk = (const float*)d_in[4];
    const float* bk = (const float*)d_in[5];
    const float* Wv = (const float*)d_in[6];
    const float* bv = (const float*)d_in[7];
    const float* Wo = (const float*)d_in[8];
    const float* bo = (const float*)d_in[9];

    char* ws = (char*)d_ws;
    auto alloc = [&](size_t bytes) {
        char* p = ws;
        ws += (bytes + 255) & ~(size_t)255;
        return p;
    };
    bf16* Xb  = (bf16*)alloc((size_t)MTOK * HID * 2);
    bf16* WqT = (bf16*)alloc((size_t)HID * HID * 2);
    bf16* WkT = (bf16*)alloc((size_t)KVDIM * HID * 2);
    bf16* WvT = (bf16*)alloc((size_t)KVDIM * HID * 2);
    bf16* WoT = (bf16*)alloc((size_t)HID * HID * 2);
    bf16* Qr  = (bf16*)alloc((size_t)NB * NHEADS * SEQ * HDIM * 2);
    bf16* Kr  = (bf16*)alloc((size_t)NB * NKVH * SEQ * HDIM * 2);
    bf16* Vr  = (bf16*)alloc((size_t)NB * NKVH * SEQ * HDIM * 2);
    bf16* Vt  = (bf16*)alloc((size_t)NB * NKVH * SEQ * HDIM * 2);
    bf16* AO  = (bf16*)alloc((size_t)MTOK * HID * 2);

    const int nX = MTOK * HID;
    f32_to_bf16_kernel<<<nX / 1024, 256, 0, stream>>>(hs, Xb, nX);
    transpose_w_kernel<<<dim3(HID / 32, HID / 32), dim3(32, 8), 0, stream>>>(Wq, WqT, HID, HID);
    transpose_w_kernel<<<dim3(KVDIM / 32, HID / 32), dim3(32, 8), 0, stream>>>(Wk, WkT, HID, KVDIM);
    transpose_w_kernel<<<dim3(KVDIM / 32, HID / 32), dim3(32, 8), 0, stream>>>(Wv, WvT, HID, KVDIM);
    transpose_w_kernel<<<dim3(HID / 32, HID / 32), dim3(32, 8), 0, stream>>>(Wo, WoT, HID, HID);
    gemm_bf16_wmma<0><<<dim3(HID / 128, MTOK / 128), 256, 0, stream>>>(
        Xb, WqT, bq, Qr, MTOK, HID, HID, NHEADS);
    gemm_bf16_wmma<0><<<dim3(KVDIM / 128, MTOK / 128), 256, 0, stream>>>(
        Xb, WkT, bk, Kr, MTOK, KVDIM, HID, NKVH);
    gemm_bf16_wmma<0><<<dim3(KVDIM / 128, MTOK / 128), 256, 0, stream>>>(
        Xb, WvT, bv, Vr, MTOK, KVDIM, HID, NKVH);
    rope_kernel<<<(NB * NHEADS * SEQ * 64) / 256, 256, 0, stream>>>(Qr, NB * NHEADS * SEQ * 64);
    rope_kernel<<<(NB * NKVH * SEQ * 64) / 256, 256, 0, stream>>>(Kr, NB * NKVH * SEQ * 64);
    transpose_v_kernel<<<dim3(HDIM / 32, SEQ / 32, NB * NKVH), dim3(32, 8), 0, stream>>>(Vr, Vt);
    flash_attn_kernel<<<dim3(SEQ / 64, NHEADS, NB), 128, 0, stream>>>(Qr, Kr, Vt, AO);
    gemm_bf16_wmma<1><<<dim3(HID / 128, MTOK / 128), 256, 0, stream>>>(
        AO, WoT, bo, d_out, MTOK, HID, HID, 1);
}
